// MambaBlock_38482906972657
// MI455X (gfx1250) — compile-verified, matrix operand order checked by reading, not run on hardware
//
#include <hip/hip_runtime.h>
#include <hip/hip_bf16.h>
#include <stdint.h>

// ---------------------------------------------------------------------------
// Mamba block for MI455X (gfx1250), wave32.
// - bf16 WMMA (v_wmma_f32_16x16x32_bf16) for the three projections
// - async global->LDS staging (ASYNCcnt) + double buffering for big GEMMs
// - non-temporal streaming for the 536MB h/h_new traffic (HBM roofline stage)
// Shapes fixed by reference: B=2, L=1024, D=1024, ED=2048, N=16, R=32, K=3
// ---------------------------------------------------------------------------

typedef __bf16 bf16_t;
typedef __attribute__((ext_vector_type(16))) __bf16 v16bf;
typedef __attribute__((ext_vector_type(8)))  __bf16 v8bf;
typedef __attribute__((ext_vector_type(8)))  float  v8f;
typedef __attribute__((ext_vector_type(4)))  float  v4f;

#define B_  2
#define L_  1024
#define Dm_ 1024
#define ED_ 2048
#define Ns_ 16
#define Rr_ 32
#define BL_ (B_ * L_)

// float -> bf16 round-to-nearest-even
__device__ __forceinline__ bf16_t f2bf(float f) {
    unsigned u = __builtin_bit_cast(unsigned, f);
    unsigned r = u + 0x7FFFu + ((u >> 16) & 1u);
    unsigned short h = (unsigned short)(r >> 16);
    return __builtin_bit_cast(bf16_t, h);
}

__global__ __launch_bounds__(256) void k_f32_to_bf16(const float* __restrict__ src,
                                                     bf16_t* __restrict__ dst, int n) {
    int i = blockIdx.x * blockDim.x + threadIdx.x;
    if (i < n) dst[i] = f2bf(src[i]);
}

// --- CDNA5 async copy helpers (inline asm; portable across toolchains) -----
__device__ __forceinline__ void async_b128(unsigned ldsOff, const bf16_t* g) {
    // copy 16 bytes from global g to LDS offset ldsOff, tracked by ASYNCcnt
    asm volatile("global_load_async_to_lds_b128 %0, %1, off"
                 :: "v"(ldsOff), "v"(g) : "memory");
}
__device__ __forceinline__ void wait_async0() {
    asm volatile("s_wait_asynccnt 0" ::: "memory");
}

__device__ __forceinline__ v16bf combine(v8bf lo, v8bf hi) {
    return __builtin_shufflevector(lo, hi, 0, 1, 2, 3, 4, 5, 6, 7,
                                          8, 9, 10, 11, 12, 13, 14, 15);
}

// Fragment from an LDS tile with row stride 40 bf16 (80B; 16B pad vs 64B row).
// A-layout (also B-layout for row-major W[N,K]): lane l -> row base+(l&15),
// K-runs of 8 at col (l>>4)*8 and 16+(l>>4)*8.
__device__ __forceinline__ v16bf lds_frag(const bf16_t (*__restrict__ tile)[40],
                                          int row, int lane) {
    const bf16_t* q = &tile[row + (lane & 15)][(lane >> 4) << 3];
    v8bf lo = *(const v8bf*)(q);
    v8bf hi = *(const v8bf*)(q + 16);
    return combine(lo, hi);
}

// Fragment directly from global row-major bf16 (used by the tiny N=64 GEMM).
__device__ __forceinline__ v16bf load_frag(const bf16_t* __restrict__ p, int ld,
                                           int row, int k0, int lane) {
    const bf16_t* q = p + (size_t)(row + (lane & 15)) * ld + k0 + ((lane >> 4) << 3);
    return combine(*(const v8bf*)(q), *(const v8bf*)(q + 16));
}

// ---------------------------------------------------------------------------
// Big GEMM: C[M,N] = A[M,K] x W[N,K]^T, bf16 in / f32 out.
// Block 256 threads = 8 waves; block tile 128(M) x 128(N); wave tile 32x64.
// K-steps of 32 staged in LDS via async b128 copies, double buffered.
// Requires M%128==0, N%128==0, K%32==0 (true for in_proj / out_proj).
// ---------------------------------------------------------------------------
__global__ __launch_bounds__(256) void k_wmma_gemm_lds(float* __restrict__ C,
                                                       const bf16_t* __restrict__ A,
                                                       const bf16_t* __restrict__ W,
                                                       int M, int N, int K) {
    __shared__ __align__(16) bf16_t sA[2][128][40];  // 128x32 tile + pad
    __shared__ __align__(16) bf16_t sW[2][128][40];

    const int tid  = threadIdx.x;
    const int lane = tid & 31;
    const int wid  = tid >> 5;
    const int blockRow = blockIdx.y * 128;
    const int blockCol = blockIdx.x * 128;
    const int waveRow  = (wid >> 1) * 32;  // 4 waves over M
    const int waveCol  = (wid & 1) * 64;   // 2 waves over N

    // Each thread copies two 16B chunks of each tile (512 chunks / tile).
    const int r0 = tid >> 2, c0 = (tid & 3) << 3;          // chunk tid
    const int r1 = (tid + 256) >> 2, c1 = c0;              // chunk tid+256 (same cc)

    auto issue = [&](int buf, int k0) {
        async_b128((unsigned)(uintptr_t)&sA[buf][r0][c0],
                   A + (size_t)(blockRow + r0) * K + k0 + c0);
        async_b128((unsigned)(uintptr_t)&sA[buf][r1][c1],
                   A + (size_t)(blockRow + r1) * K + k0 + c1);
        async_b128((unsigned)(uintptr_t)&sW[buf][r0][c0],
                   W + (size_t)(blockCol + r0) * K + k0 + c0);
        async_b128((unsigned)(uintptr_t)&sW[buf][r1][c1],
                   W + (size_t)(blockCol + r1) * K + k0 + c1);
    };

    v8f acc[2][4] = {};

    int buf = 0;
    issue(0, 0);
    wait_async0();
    __syncthreads();

    for (int k0 = 0; k0 < K; k0 += 32) {
        if (k0 + 32 < K) issue(buf ^ 1, k0 + 32);

        v16bf a0 = lds_frag(sA[buf], waveRow, lane);
        v16bf a1 = lds_frag(sA[buf], waveRow + 16, lane);
        v16bf b0 = lds_frag(sW[buf], waveCol, lane);
        v16bf b1 = lds_frag(sW[buf], waveCol + 16, lane);
        v16bf b2 = lds_frag(sW[buf], waveCol + 32, lane);
        v16bf b3 = lds_frag(sW[buf], waveCol + 48, lane);

        acc[0][0] = __builtin_amdgcn_wmma_f32_16x16x32_bf16(false, a0, false, b0, (short)0, acc[0][0], false, false);
        acc[0][1] = __builtin_amdgcn_wmma_f32_16x16x32_bf16(false, a0, false, b1, (short)0, acc[0][1], false, false);
        acc[0][2] = __builtin_amdgcn_wmma_f32_16x16x32_bf16(false, a0, false, b2, (short)0, acc[0][2], false, false);
        acc[0][3] = __builtin_amdgcn_wmma_f32_16x16x32_bf16(false, a0, false, b3, (short)0, acc[0][3], false, false);
        acc[1][0] = __builtin_amdgcn_wmma_f32_16x16x32_bf16(false, a1, false, b0, (short)0, acc[1][0], false, false);
        acc[1][1] = __builtin_amdgcn_wmma_f32_16x16x32_bf16(false, a1, false, b1, (short)0, acc[1][1], false, false);
        acc[1][2] = __builtin_amdgcn_wmma_f32_16x16x32_bf16(false, a1, false, b2, (short)0, acc[1][2], false, false);
        acc[1][3] = __builtin_amdgcn_wmma_f32_16x16x32_bf16(false, a1, false, b3, (short)0, acc[1][3], false, false);

        wait_async0();       // next buffer landed
        __syncthreads();     // everyone done reading current buffer
        buf ^= 1;
    }

    // C/D layout: lane l holds col (l&15); VGPR r holds row r + 8*(l>>4).
    const int mBase = blockRow + waveRow + ((lane >> 4) << 3);
    const int nBase = blockCol + waveCol + (lane & 15);
#pragma unroll
    for (int mi = 0; mi < 2; ++mi)
#pragma unroll
        for (int ni = 0; ni < 4; ++ni)
#pragma unroll
            for (int r = 0; r < 8; ++r)
                C[(size_t)(mBase + mi * 16 + r) * N + nBase + ni * 16] = acc[mi][ni][r];
}

// ---------------------------------------------------------------------------
// Small GEMM (N=64): direct-from-global version. Block tile 128x64, wave 32x32.
// ---------------------------------------------------------------------------
__global__ __launch_bounds__(256) void k_wmma_gemm_bt(float* __restrict__ C,
                                                      const bf16_t* __restrict__ A,
                                                      const bf16_t* __restrict__ W,
                                                      int M, int N, int K) {
    const int lane  = threadIdx.x & 31;
    const int wid   = threadIdx.x >> 5;
    const int rowBase = blockIdx.y * 128 + (wid >> 1) * 32;
    const int colBase = blockIdx.x * 64  + (wid & 1)  * 32;

    v8f acc00 = {}, acc01 = {}, acc10 = {}, acc11 = {};

    for (int k0 = 0; k0 < K; k0 += 32) {
        v16bf a0 = load_frag(A, K, rowBase,      k0, lane);
        v16bf a1 = load_frag(A, K, rowBase + 16, k0, lane);
        v16bf b0 = load_frag(W, K, colBase,      k0, lane);
        v16bf b1 = load_frag(W, K, colBase + 16, k0, lane);
        if (k0 + 64 <= K) {  // near-scope prefetch of next k-step
            __builtin_prefetch(A + (size_t)(rowBase + (lane & 15)) * K + k0 + 32, 0, 3);
            __builtin_prefetch(W + (size_t)(colBase + (lane & 15)) * K + k0 + 32, 0, 3);
        }
        acc00 = __builtin_amdgcn_wmma_f32_16x16x32_bf16(false, a0, false, b0, (short)0, acc00, false, false);
        acc01 = __builtin_amdgcn_wmma_f32_16x16x32_bf16(false, a0, false, b1, (short)0, acc01, false, false);
        acc10 = __builtin_amdgcn_wmma_f32_16x16x32_bf16(false, a1, false, b0, (short)0, acc10, false, false);
        acc11 = __builtin_amdgcn_wmma_f32_16x16x32_bf16(false, a1, false, b1, (short)0, acc11, false, false);
    }

    const int n0 = colBase + (lane & 15);
    const int m0 = rowBase + ((lane >> 4) << 3);
#pragma unroll
    for (int r = 0; r < 8; ++r) {
        C[(size_t)(m0 + r) * N + n0]           = acc00[r];
        C[(size_t)(m0 + r) * N + n0 + 16]      = acc01[r];
        C[(size_t)(m0 + r + 16) * N + n0]      = acc10[r];
        C[(size_t)(m0 + r + 16) * N + n0 + 16] = acc11[r];
    }
}

// Depthwise conv1d (K=3, 'same') + bias + SiLU; fp32 + bf16 outputs.
__global__ __launch_bounds__(256) void k_conv_silu(const float* __restrict__ xz,
                                                   const float* __restrict__ cw,
                                                   const float* __restrict__ cb,
                                                   float* __restrict__ xs,
                                                   bf16_t* __restrict__ xsb) {
    int idx = blockIdx.x * blockDim.x + threadIdx.x;
    if (idx >= BL_ * ED_) return;
    int e  = idx & (ED_ - 1);
    int bl = idx >> 11;
    int l  = bl & (L_ - 1);
    float w0 = cw[e * 3 + 0], w1 = cw[e * 3 + 1], w2 = cw[e * 3 + 2];
    float acc = cb[e] + w1 * xz[(size_t)bl * (2 * ED_) + e];
    if (l > 0)      acc += w0 * xz[(size_t)(bl - 1) * (2 * ED_) + e];
    if (l < L_ - 1) acc += w2 * xz[(size_t)(bl + 1) * (2 * ED_) + e];
    float s = acc / (1.0f + expf(-acc));
    xs[idx]  = s;
    xsb[idx] = f2bf(s);
}

// delta = softplus(x_dbl[:, :R] @ dt_proj_w^T + b); K=32 scalar fp32 dot.
__global__ __launch_bounds__(256) void k_delta(const float* __restrict__ xdbl,
                                               const float* __restrict__ dtw,
                                               const float* __restrict__ dtb,
                                               float* __restrict__ delta) {
    __shared__ float sd[Rr_];
    int bl = blockIdx.y;
    int e  = blockIdx.x * 256 + threadIdx.x;
    if (threadIdx.x < Rr_) sd[threadIdx.x] = xdbl[bl * 64 + threadIdx.x];
    __syncthreads();
    float acc = dtb[e];
#pragma unroll
    for (int r = 0; r < Rr_; ++r) acc += sd[r] * dtw[e * Rr_ + r];
    float d = (acc > 20.f) ? acc : log1pf(expf(acc));
    delta[(size_t)bl * ED_ + e] = d;
}

// Fused SSM: h_new = exp(delta*A)*h + delta*B*xs ; y = h_new.C + D*xs ;
// yg = y*silu(z). h / h_new streamed non-temporally (no reuse; spare the L2).
__global__ __launch_bounds__(256) void k_ssm(const float* __restrict__ h,
                                             const float* __restrict__ xdbl,
                                             const float* __restrict__ delta,
                                             const float* __restrict__ xs,
                                             const float* __restrict__ xz,
                                             const float* __restrict__ alog,
                                             const float* __restrict__ Dv,
                                             float* __restrict__ h_new,
                                             bf16_t* __restrict__ yg) {
    __shared__ float sBC[2 * Ns_];
    int idx = blockIdx.x * 256 + threadIdx.x;
    int e  = idx & (ED_ - 1);
    int bl = idx >> 11;
    if (threadIdx.x < 2 * Ns_)
        sBC[threadIdx.x] = xdbl[bl * 64 + Rr_ + threadIdx.x];
    __syncthreads();

    float dlt = delta[idx];
    float xsv = xs[idx];
    float zv  = xz[(size_t)bl * (2 * ED_) + ED_ + e];

    const v4f* h4  = (const v4f*)(h     + (size_t)idx * Ns_);
    v4f*       hn4 = (v4f*)      (h_new + (size_t)idx * Ns_);
    float y = 0.f;
#pragma unroll
    for (int i = 0; i < 4; ++i) {
        v4f hv = __builtin_nontemporal_load(&h4[i]);
        v4f hout;
#pragma unroll
        for (int j = 0; j < 4; ++j) {
            int n = i * 4 + j;
            float a  = -expf(alog[e * Ns_ + n]);
            float da = expf(dlt * a);
            float v  = da * hv[j] + dlt * sBC[n] * xsv;
            hout[j] = v;
            y += v * sBC[Ns_ + n];
        }
        __builtin_nontemporal_store(hout, &hn4[i]);
    }
    y += Dv[e] * xsv;
    float g = zv / (1.f + expf(-zv));
    yg[idx] = f2bf(y * g);
}

// ---------------------------------------------------------------------------
extern "C" void kernel_launch(void* const* d_in, const int* in_sizes, int n_in,
                              void* d_out, int out_size, void* d_ws, size_t ws_size,
                              hipStream_t stream) {
    (void)in_sizes; (void)n_in; (void)out_size; (void)ws_size;
    const float* x     = (const float*)d_in[0];
    const float* h     = (const float*)d_in[1];
    const float* ipw   = (const float*)d_in[2];
    const float* convw = (const float*)d_in[3];
    const float* convb = (const float*)d_in[4];
    const float* xpw   = (const float*)d_in[5];
    const float* dtw   = (const float*)d_in[6];
    const float* dtb   = (const float*)d_in[7];
    const float* alog  = (const float*)d_in[8];
    const float* Dv    = (const float*)d_in[9];
    const float* opw   = (const float*)d_in[10];

    float* out   = (float*)d_out;
    float* h_new = out + (size_t)BL_ * Dm_;

    char* ws = (char*)d_ws;
    auto alloc = [&](size_t bytes) {
        char* p = ws;
        ws += (bytes + 255) & ~(size_t)255;
        return p;
    };
    bf16_t* x_bf   = (bf16_t*)alloc((size_t)BL_ * Dm_ * 2);
    bf16_t* ipw_bf = (bf16_t*)alloc((size_t)2 * ED_ * Dm_ * 2);
    bf16_t* xpw_bf = (bf16_t*)alloc((size_t)64 * ED_ * 2);
    bf16_t* opw_bf = (bf16_t*)alloc((size_t)Dm_ * ED_ * 2);
    float*  xz     = (float*) alloc((size_t)BL_ * 2 * ED_ * 4);
    float*  xs     = (float*) alloc((size_t)BL_ * ED_ * 4);
    bf16_t* xs_bf  = (bf16_t*)alloc((size_t)BL_ * ED_ * 2);
    float*  xdbl   = (float*) alloc((size_t)BL_ * 64 * 4);
    float*  dlt    = (float*) alloc((size_t)BL_ * ED_ * 4);
    bf16_t* yg_bf  = (bf16_t*)alloc((size_t)BL_ * ED_ * 2);

    auto cvt = [&](const float* s, bf16_t* d, int n) {
        k_f32_to_bf16<<<(n + 255) / 256, 256, 0, stream>>>(s, d, n);
    };
    cvt(x,   x_bf,   BL_ * Dm_);
    cvt(ipw, ipw_bf, 2 * ED_ * Dm_);
    cvt(xpw, xpw_bf, 64 * ED_);
    cvt(opw, opw_bf, Dm_ * ED_);

    // xz = x @ in_proj_w^T : M=2048, N=4096, K=1024 (async-LDS WMMA)
    k_wmma_gemm_lds<<<dim3((2 * ED_) / 128, BL_ / 128), 256, 0, stream>>>(
        xz, x_bf, ipw_bf, BL_, 2 * ED_, Dm_);

    k_conv_silu<<<(BL_ * ED_ + 255) / 256, 256, 0, stream>>>(xz, convw, convb, xs, xs_bf);

    // x_dbl = xs @ x_proj_w^T : M=2048, N=64, K=2048 (small; direct WMMA)
    k_wmma_gemm_bt<<<dim3(1, BL_ / 128), 256, 0, stream>>>(
        xdbl, xs_bf, xpw_bf, BL_, 64, ED_);

    k_delta<<<dim3(ED_ / 256, BL_), 256, 0, stream>>>(xdbl, dtw, dtb, dlt);

    k_ssm<<<(BL_ * ED_) / 256, 256, 0, stream>>>(h, xdbl, dlt, xs, xz, alog, Dv,
                                                 h_new, yg_bf);

    // out = yg @ out_proj_w^T : M=2048, N=1024, K=2048 (async-LDS WMMA)
    k_wmma_gemm_lds<<<dim3(Dm_ / 128, BL_ / 128), 256, 0, stream>>>(
        out, yg_bf, opw_bf, BL_, Dm_, ED_);
}
